// Simple_Moe_40810779246876
// MI455X (gfx1250) — compile-verified, matrix operand order checked by reading, not run on hardware
//
#include <hip/hip_runtime.h>
#include <hip/hip_bf16.h>

// Problem constants (match reference)
#define B_TOK     4096
#define DIN_DIM   512
#define F_DIM     512
#define H_DIM     2048
#define O_DIM     512
#define E_NUM     8
#define TEMP_F    1.5f
#define MIN_NOISE 0.01f

// Routed-row capacity: exactly B*K real rows + per-expert pad to 64-row tiles
#define MAX_ROWS  (B_TOK * 2 + E_NUM * 64)   // 8704
#define ROW_TILES (MAX_ROWS / 64)            // 136

// LDS tile geometry for the bf16 expert GEMMs: 64 rows x 64 K, stride 72 (bank pad)
#define LSTRIDE   72
#define LBUF      (64 * LSTRIDE)             // ushorts per matrix per buffer

typedef __attribute__((ext_vector_type(16))) __bf16          v16bf;
typedef __attribute__((ext_vector_type(8)))  float           v8f;
typedef __attribute__((ext_vector_type(2)))  float           v2f;
typedef __attribute__((ext_vector_type(8)))  unsigned short  v8u;

// f32 -> bf16, round-to-nearest-even
__device__ __forceinline__ unsigned short f2bf(float x) {
  unsigned u = __float_as_uint(x);
  u += 0x7FFFu + ((u >> 16) & 1u);
  return (unsigned short)(u >> 16);
}

// Async global->LDS copy of 8 bytes (GLOBAL_LOAD_ASYNC_TO_LDS_B64, ASYNCcnt).
// lds_addr = wave-relative LDS byte offset (low 32 bits of generic shared ptr).
__device__ __forceinline__ void async_b64(unsigned lds_addr, const void* gptr) {
  unsigned long long ga = (unsigned long long)(uintptr_t)gptr;
  asm volatile("global_load_async_to_lds_b64 %0, %1, off"
               :: "v"(lds_addr), "v"(ga)
               : "memory");
}
__device__ __forceinline__ void wait_async() {
  asm volatile("s_wait_asynccnt 0x0" ::: "memory");
}
__device__ __forceinline__ unsigned lds_off(const void* p) {
  return (unsigned)(uintptr_t)p;
}

// Load a 16x16x32-bf16 WMMA fragment from an LDS tile stored row-major
// [rows][64 bf16] with row stride LSTRIDE. Per ISA 16-bit A layout:
// lane = row(mod16), half = lane>>4 selects K {h*8..h*8+7} U {16+h*8..+7};
// ks selects which 32-wide K step of the 64-wide stage.
__device__ __forceinline__ v16bf frag64(const unsigned short* lds, int row, int half, int ks) {
  const unsigned short* p = lds + row * LSTRIDE + ks * 32 + half * 8;
  union { v16bf v; v8u u[2]; } f;
  f.u[0] = *(const v8u*)(p);
  f.u[1] = *(const v8u*)(p + 16);
  return f.v;
}

// ---------------------------------------------------------------- init
__global__ void k_init(float* __restrict__ out, int* __restrict__ counts) {
  int i = blockIdx.x * blockDim.x + threadIdx.x;
  if (i < B_TOK * O_DIM) out[i] = 0.0f;
  if (i < E_NUM) counts[i] = 0;
}

// ---------------------------------------------------------------- weight convert+transpose
// Wt[e][n][k] (bf16) = W[e][k][n] (f32);  grid: (Nd/64, Kd/64, E)
__global__ __launch_bounds__(256) void k_wconv(const float* __restrict__ W,
                                               unsigned short* __restrict__ Wt,
                                               int Kd, int Nd) {
  __shared__ unsigned short s[64 * 65];
  const int e = blockIdx.z;
  const float* Wsrc = W + (size_t)e * Kd * Nd;
  unsigned short* Wdst = Wt + (size_t)e * Kd * Nd;
  const int n0 = blockIdx.x * 64, k0 = blockIdx.y * 64;
  const int tid = threadIdx.x;
#pragma unroll
  for (int i = 0; i < 16; ++i) {           // read coalesced along n
    int idx = tid + i * 256;
    int k = idx >> 6, n = idx & 63;
    s[k * 65 + n] = f2bf(Wsrc[(size_t)(k0 + k) * Nd + n0 + n]);
  }
  __syncthreads();
#pragma unroll
  for (int i = 0; i < 16; ++i) {           // write coalesced along k
    int idx = tid + i * 256;
    int n = idx >> 6, k = idx & 63;
    Wdst[(size_t)(n0 + n) * Kd + k0 + k] = s[k * 65 + n];
  }
}

// ---------------------------------------------------------------- backbone (f32 WMMA, exact)
// features[B,F] = x[B,DIN] @ Wb[DIN,F] + bb ; write f32 (gate) + bf16 (experts)
__global__ __launch_bounds__(256) void k_backbone(
    const float* __restrict__ x, const float* __restrict__ Wb, const float* __restrict__ bb,
    float* __restrict__ feat_f32, unsigned short* __restrict__ feat_bf16) {
  __shared__ float sA[64 * 34];
  __shared__ float sB[64 * 34];
  const int tid = threadIdx.x;
  const int lane = tid & 31, wave = tid >> 5;
  const int row0 = blockIdx.y * 64;
  const int col0 = blockIdx.x * 64;
  const int wm = wave >> 1;
  const int nb0 = (wave & 1) * 32;
  const int ml = lane & 15, half = lane >> 4;

  v8f c0 = {}; v8f c1 = {};
  for (int k0 = 0; k0 < DIN_DIM; k0 += 32) {
#pragma unroll
    for (int i = 0; i < 8; ++i) {
      int idx = tid + i * 256;
      int r = idx >> 5, k = idx & 31;
      sA[r * 34 + k] = x[(row0 + r) * DIN_DIM + k0 + k];
    }
#pragma unroll
    for (int i = 0; i < 8; ++i) {
      int idx = tid + i * 256;
      int k = idx >> 6, n = idx & 63;
      sB[n * 34 + k] = Wb[(k0 + k) * F_DIM + col0 + n];
    }
    __syncthreads();
#pragma unroll
    for (int kk = 0; kk < 8; ++kk) {
      int kb = kk * 4 + 2 * half;
      v2f a  = *(const v2f*)(sA + (wm * 16 + ml) * 34 + kb);
      v2f b0 = *(const v2f*)(sB + (nb0 + ml) * 34 + kb);
      v2f b1 = *(const v2f*)(sB + (nb0 + 16 + ml) * 34 + kb);
      c0 = __builtin_amdgcn_wmma_f32_16x16x4_f32(false, a, false, b0, (short)0, c0, false, false);
      c1 = __builtin_amdgcn_wmma_f32_16x16x4_f32(false, a, false, b1, (short)0, c1, false, false);
    }
    __syncthreads();
  }
#pragma unroll
  for (int v = 0; v < 8; ++v) {
    int m = half * 8 + v;
    int gr = row0 + wm * 16 + m;
    int gc0 = col0 + nb0 + ml;
    int gc1 = gc0 + 16;
    float f0 = c0[v] + bb[gc0];
    float f1 = c1[v] + bb[gc1];
    feat_f32[gr * F_DIM + gc0] = f0;
    feat_f32[gr * F_DIM + gc1] = f1;
    feat_bf16[gr * F_DIM + gc0] = f2bf(f0);
    feat_bf16[gr * F_DIM + gc1] = f2bf(f1);
  }
}

// ---------------------------------------------------------------- gating (one wave32 per token)
__global__ __launch_bounds__(256) void k_gate(
    const float* __restrict__ feat, const float* __restrict__ Wg, const float* __restrict__ bg,
    const float* __restrict__ Wn, const float* __restrict__ bn, const float* __restrict__ noise,
    int* __restrict__ counts, int* __restrict__ bucket_tok, float* __restrict__ bucket_w) {
  const int lane = threadIdx.x & 31, wave = threadIdx.x >> 5;
  const int b = blockIdx.x * 8 + wave;
  float ag[E_NUM], an[E_NUM];
#pragma unroll
  for (int e = 0; e < E_NUM; ++e) { ag[e] = 0.f; an[e] = 0.f; }
  for (int f = lane; f < F_DIM; f += 32) {
    float fv = feat[b * F_DIM + f];
#pragma unroll
    for (int e = 0; e < E_NUM; ++e) {
      ag[e] += fv * Wg[f * E_NUM + e];
      an[e] += fv * Wn[f * E_NUM + e];
    }
  }
#pragma unroll
  for (int e = 0; e < E_NUM; ++e)
    for (int m = 16; m >= 1; m >>= 1) {
      ag[e] += __shfl_xor(ag[e], m, 32);
      an[e] += __shfl_xor(an[e], m, 32);
    }
  if (lane == 0) {
    float g[E_NUM];
#pragma unroll
    for (int e = 0; e < E_NUM; ++e) {
      float raw = ag[e] + bg[e];
      float xn = an[e] + bn[e];
      float sp = (xn > 20.f) ? xn : log1pf(__expf(xn));     // softplus
      g[e] = raw + (sp + MIN_NOISE) * noise[b * E_NUM + e];
    }
    int i0 = 0;
#pragma unroll
    for (int e = 1; e < E_NUM; ++e) if (g[e] > g[i0]) i0 = e;   // first index wins ties
    int i1 = (i0 == 0) ? 1 : 0;
#pragma unroll
    for (int e = 0; e < E_NUM; ++e) if (e != i0 && g[e] > g[i1]) i1 = e;
    float s1 = __expf((g[i1] - g[i0]) / TEMP_F);
    float inv = 1.0f / (1.0f + s1);
    float w0 = inv, w1 = s1 * inv;
    int p0 = atomicAdd(&counts[i0], 1);
    bucket_tok[i0 * B_TOK + p0] = b;  bucket_w[i0 * B_TOK + p0] = w0;
    int p1 = atomicAdd(&counts[i1], 1);
    bucket_tok[i1 * B_TOK + p1] = b;  bucket_w[i1 * B_TOK + p1] = w1;
  }
}

// ---------------------------------------------------------------- 64-aligned exclusive offsets
__global__ void k_offsets(const int* __restrict__ counts, int* __restrict__ offsets,
                          int* __restrict__ padded_total) {
  if (threadIdx.x == 0 && blockIdx.x == 0) {
    int off = 0;
    for (int e = 0; e < E_NUM; ++e) {
      offsets[e] = off;
      off += ((counts[e] + 63) >> 6) << 6;
    }
    *padded_total = off;
  }
}

// ---------------------------------------------------------------- compact routed rows (pad w=0)
__global__ __launch_bounds__(256) void k_compact(
    const int* __restrict__ counts, const int* __restrict__ offsets,
    const int* __restrict__ bucket_tok, const float* __restrict__ bucket_w,
    int* __restrict__ row_token, float* __restrict__ row_w, int* __restrict__ row_expert) {
  const int e = blockIdx.x;
  const int cnt = counts[e], base = offsets[e];
  const int padded = ((cnt + 63) >> 6) << 6;
  for (int i = threadIdx.x; i < padded; i += 256) {
    int tok = 0; float w = 0.f;
    if (i < cnt) { tok = bucket_tok[e * B_TOK + i]; w = bucket_w[e * B_TOK + i]; }
    row_token[base + i] = tok;
    row_w[base + i] = w;
    row_expert[base + i] = e;
  }
}

// ---------------------------------------------------------------- expert layer 1 (bf16 WMMA)
// h[row, H] = relu(feat_bf16[token[row]] @ W1 + b1); A gathered, B from W1t[e][n][k].
// Double-buffered LDS stages filled with GLOBAL_LOAD_ASYNC_TO_LDS_B64.
__global__ __launch_bounds__(256) void k_expert1(
    const unsigned short* __restrict__ feat_bf16,
    const unsigned short* __restrict__ W1t, const float* __restrict__ bias1,
    const int* __restrict__ row_token, const int* __restrict__ row_expert,
    const int* __restrict__ padded_total,
    unsigned short* __restrict__ h_out) {
  __shared__ unsigned short sA[2 * LBUF];
  __shared__ unsigned short sB[2 * LBUF];
  __shared__ int s_tok[64];
  const int pt = *padded_total;
  const int row0 = blockIdx.y * 64;
  if (row0 >= pt) return;
  const int col0 = blockIdx.x * 64;
  const int tid = threadIdx.x;
  const int lane = tid & 31, wave = tid >> 5;
  const int wm = wave >> 1, nb0 = (wave & 1) * 32;
  const int ml = lane & 15, half = lane >> 4;
  const int e = row_expert[row0];
  const unsigned short* W = W1t + (size_t)e * F_DIM * H_DIM;   // [n][k], row stride F_DIM
  if (tid < 64) s_tok[tid] = row_token[row0 + tid];
  __syncthreads();

  const int cch = tid >> 4;          // chunk row 0..15 handled per i-step
  const int kc  = (tid & 15) * 4;    // 4 bf16 = 8 bytes per async op

  auto stage = [&](int buf, int k0) {
    unsigned short* bA = sA + buf * LBUF;
    unsigned short* bB = sB + buf * LBUF;
#pragma unroll
    for (int i = 0; i < 4; ++i) {
      int r = cch + i * 16;          // 0..63
      async_b64(lds_off(&bA[r * LSTRIDE + kc]),
                feat_bf16 + (size_t)s_tok[r] * F_DIM + k0 + kc);
      async_b64(lds_off(&bB[r * LSTRIDE + kc]),
                W + (size_t)(col0 + r) * F_DIM + k0 + kc);
    }
  };

  v8f c0 = {}, c1 = {};
  stage(0, 0);
  wait_async();
  __syncthreads();
  const int nk = F_DIM / 64;
  for (int i = 0; i < nk; ++i) {
    const int cur = i & 1;
    if (i + 1 < nk) stage(1 - cur, (i + 1) * 64);
    const unsigned short* bA = sA + cur * LBUF;
    const unsigned short* bB = sB + cur * LBUF;
    v16bf a0  = frag64(bA, wm * 16 + ml, half, 0);
    v16bf a1  = frag64(bA, wm * 16 + ml, half, 1);
    v16bf b00 = frag64(bB, nb0 + ml, half, 0);
    v16bf b01 = frag64(bB, nb0 + ml, half, 1);
    v16bf b10 = frag64(bB, nb0 + 16 + ml, half, 0);
    v16bf b11 = frag64(bB, nb0 + 16 + ml, half, 1);
    c0 = __builtin_amdgcn_wmma_f32_16x16x32_bf16(false, a0, false, b00, (short)0, c0, false, false);
    c1 = __builtin_amdgcn_wmma_f32_16x16x32_bf16(false, a0, false, b10, (short)0, c1, false, false);
    c0 = __builtin_amdgcn_wmma_f32_16x16x32_bf16(false, a1, false, b01, (short)0, c0, false, false);
    c1 = __builtin_amdgcn_wmma_f32_16x16x32_bf16(false, a1, false, b11, (short)0, c1, false, false);
    wait_async();
    __syncthreads();
  }
#pragma unroll
  for (int v = 0; v < 8; ++v) {
    int m = half * 8 + v;
    int gr = row0 + wm * 16 + m;
    int gc0 = col0 + nb0 + ml;
    int gc1 = gc0 + 16;
    float y0 = fmaxf(c0[v] + bias1[e * H_DIM + gc0], 0.f);
    float y1 = fmaxf(c1[v] + bias1[e * H_DIM + gc1], 0.f);
    h_out[(size_t)gr * H_DIM + gc0] = f2bf(y0);
    h_out[(size_t)gr * H_DIM + gc1] = f2bf(y1);
  }
}

// ---------------------------------------------------------------- expert layer 2 + scatter
// out[token] += w_row * (h[row] @ W2 + b2); B from W2t[e][n][k].
__global__ __launch_bounds__(256) void k_expert2(
    const unsigned short* __restrict__ h_in,
    const unsigned short* __restrict__ W2t, const float* __restrict__ bias2,
    const int* __restrict__ row_token, const float* __restrict__ row_w,
    const int* __restrict__ row_expert, const int* __restrict__ padded_total,
    float* __restrict__ out) {
  __shared__ unsigned short sA[2 * LBUF];
  __shared__ unsigned short sB[2 * LBUF];
  const int pt = *padded_total;
  const int row0 = blockIdx.y * 64;
  if (row0 >= pt) return;
  const int col0 = blockIdx.x * 64;
  const int tid = threadIdx.x;
  const int lane = tid & 31, wave = tid >> 5;
  const int wm = wave >> 1, nb0 = (wave & 1) * 32;
  const int ml = lane & 15, half = lane >> 4;
  const int e = row_expert[row0];
  const unsigned short* W = W2t + (size_t)e * H_DIM * O_DIM;   // [n][k], row stride H_DIM

  const int cch = tid >> 4;
  const int kc  = (tid & 15) * 4;

  auto stage = [&](int buf, int k0) {
    unsigned short* bA = sA + buf * LBUF;
    unsigned short* bB = sB + buf * LBUF;
#pragma unroll
    for (int i = 0; i < 4; ++i) {
      int r = cch + i * 16;
      async_b64(lds_off(&bA[r * LSTRIDE + kc]),
                h_in + (size_t)(row0 + r) * H_DIM + k0 + kc);
      async_b64(lds_off(&bB[r * LSTRIDE + kc]),
                W + (size_t)(col0 + r) * H_DIM + k0 + kc);
    }
  };

  v8f c0 = {}, c1 = {};
  stage(0, 0);
  wait_async();
  __syncthreads();
  const int nk = H_DIM / 64;
  for (int i = 0; i < nk; ++i) {
    const int cur = i & 1;
    if (i + 1 < nk) stage(1 - cur, (i + 1) * 64);
    const unsigned short* bA = sA + cur * LBUF;
    const unsigned short* bB = sB + cur * LBUF;
    v16bf a0  = frag64(bA, wm * 16 + ml, half, 0);
    v16bf a1  = frag64(bA, wm * 16 + ml, half, 1);
    v16bf b00 = frag64(bB, nb0 + ml, half, 0);
    v16bf b01 = frag64(bB, nb0 + ml, half, 1);
    v16bf b10 = frag64(bB, nb0 + 16 + ml, half, 0);
    v16bf b11 = frag64(bB, nb0 + 16 + ml, half, 1);
    c0 = __builtin_amdgcn_wmma_f32_16x16x32_bf16(false, a0, false, b00, (short)0, c0, false, false);
    c1 = __builtin_amdgcn_wmma_f32_16x16x32_bf16(false, a0, false, b10, (short)0, c1, false, false);
    c0 = __builtin_amdgcn_wmma_f32_16x16x32_bf16(false, a1, false, b01, (short)0, c0, false, false);
    c1 = __builtin_amdgcn_wmma_f32_16x16x32_bf16(false, a1, false, b11, (short)0, c1, false, false);
    wait_async();
    __syncthreads();
  }
#pragma unroll
  for (int v = 0; v < 8; ++v) {
    int m = half * 8 + v;
    int gr = row0 + wm * 16 + m;
    int tok = row_token[gr];
    float w = row_w[gr];                      // 0 for padding rows -> adds 0
    int gc0 = col0 + nb0 + ml;
    int gc1 = gc0 + 16;
    float y0 = (c0[v] + bias2[e * O_DIM + gc0]) * w;
    float y1 = (c1[v] + bias2[e * O_DIM + gc1]) * w;
    atomicAdd(&out[tok * O_DIM + gc0], y0);
    atomicAdd(&out[tok * O_DIM + gc1], y1);
  }
}

// ---------------------------------------------------------------- launch
extern "C" void kernel_launch(void* const* d_in, const int* in_sizes, int n_in,
                              void* d_out, int out_size, void* d_ws, size_t ws_size,
                              hipStream_t stream) {
  (void)in_sizes; (void)n_in; (void)out_size; (void)ws_size;
  const float* x     = (const float*)d_in[0];
  const float* noise = (const float*)d_in[1];
  const float* Wb    = (const float*)d_in[2];
  const float* bb    = (const float*)d_in[3];
  const float* Wg    = (const float*)d_in[4];
  const float* bg    = (const float*)d_in[5];
  const float* Wn    = (const float*)d_in[6];
  const float* bn    = (const float*)d_in[7];
  const float* W1    = (const float*)d_in[8];
  const float* b1    = (const float*)d_in[9];
  const float* W2    = (const float*)d_in[10];
  const float* b2    = (const float*)d_in[11];
  float* out = (float*)d_out;

  char* ws = (char*)d_ws;
  size_t off = 0;
  auto alloc = [&](size_t bytes) -> void* {
    off = (off + 255) & ~(size_t)255;
    void* p = ws + off;
    off += bytes;
    return p;
  };
  float*          feat_f32   = (float*)         alloc((size_t)B_TOK * F_DIM * 4);
  unsigned short* feat_bf16  = (unsigned short*)alloc((size_t)B_TOK * F_DIM * 2);
  unsigned short* h_ws       = (unsigned short*)alloc((size_t)MAX_ROWS * H_DIM * 2);
  unsigned short* W1t        = (unsigned short*)alloc((size_t)E_NUM * F_DIM * H_DIM * 2);
  unsigned short* W2t        = (unsigned short*)alloc((size_t)E_NUM * H_DIM * O_DIM * 2);
  int*            bucket_tok = (int*)           alloc((size_t)E_NUM * B_TOK * 4);
  float*          bucket_w   = (float*)         alloc((size_t)E_NUM * B_TOK * 4);
  int*            row_token  = (int*)           alloc((size_t)MAX_ROWS * 4);
  float*          row_w      = (float*)         alloc((size_t)MAX_ROWS * 4);
  int*            row_expert = (int*)           alloc((size_t)MAX_ROWS * 4);
  int*            counts     = (int*)           alloc(64);
  int*            offsets    = (int*)           alloc(64);
  int*            padded_tot = (int*)           alloc(64);

  k_init<<<(B_TOK * O_DIM + 255) / 256, 256, 0, stream>>>(out, counts);
  k_backbone<<<dim3(F_DIM / 64, B_TOK / 64), 256, 0, stream>>>(x, Wb, bb, feat_f32, feat_bf16);
  k_wconv<<<dim3(H_DIM / 64, F_DIM / 64, E_NUM), 256, 0, stream>>>(W1, W1t, F_DIM, H_DIM);
  k_wconv<<<dim3(O_DIM / 64, H_DIM / 64, E_NUM), 256, 0, stream>>>(W2, W2t, H_DIM, O_DIM);
  k_gate<<<B_TOK / 8, 256, 0, stream>>>(feat_f32, Wg, bg, Wn, bn, noise,
                                        counts, bucket_tok, bucket_w);
  k_offsets<<<1, 32, 0, stream>>>(counts, offsets, padded_tot);
  k_compact<<<E_NUM, 256, 0, stream>>>(counts, offsets, bucket_tok, bucket_w,
                                       row_token, row_w, row_expert);
  k_expert1<<<dim3(H_DIM / 64, ROW_TILES), 256, 0, stream>>>(
      feat_bf16, W1t, b1, row_token, row_expert, padded_tot, h_ws);
  k_expert2<<<dim3(O_DIM / 64, ROW_TILES), 256, 0, stream>>>(
      h_ws, W2t, b2, row_token, row_w, row_expert, padded_tot, out);
}
